// GIN_50663434223942
// MI455X (gfx1250) — matrix operand order repaired and checked by reading, not compiled, not run on hardware
//
#include <hip/hip_runtime.h>
#include <stdint.h>

#define N_NODES  100000
#define N_EDGES  1600000
#define N_GRAPHS 2048
#define HID      128
#define N_LAYERS 3

typedef __attribute__((ext_vector_type(16))) __bf16 v16bf;
typedef __attribute__((ext_vector_type(8)))  float  v8f;

union Frag16 { uint32_t u[8]; uint4 q[2]; v16bf v; };

__device__ __forceinline__ uint32_t f2bf(float f) {
  uint32_t u = __builtin_bit_cast(uint32_t, f);
  return (u + 0x7FFFu + ((u >> 16) & 1u)) >> 16;   // round-to-nearest-even
}
__device__ __forceinline__ uint32_t pack2bf(float lo, float hi) {
  return f2bf(lo) | (f2bf(hi) << 16);
}

// Pack f32 weight W[K x 128] row-major into per-lane-contiguous B fragments:
// layout [K/32 kc][8 w][32 lane][8 v] dwords, so each lane's fragment is one
// contiguous 32B run (two global_load_b128).
// Fragment semantics (B 32x16, this code's contiguous k order, same as A): lane = half*16 + (n%16),
// dword v holds K = kc*32 + half*16 + {2v, 2v+1} of column n = w*16 + lane%16.
__global__ void pack_w_kernel(const float* __restrict__ W, uint32_t* __restrict__ Bp, int K) {
  int tid = blockIdx.x * blockDim.x + threadIdx.x;
  int total = (K >> 1) * HID;
  if (tid >= total) return;
  int v    = tid & 7;
  int lane = (tid >> 3) & 31;
  int w    = (tid >> 8) & 7;
  int kc   = tid >> 11;
  int n = w * 16 + (lane & 15);
  int k = kc * 32 + ((lane >> 4) << 4) + 2 * v;
  Bp[tid] = pack2bf(W[k * HID + n], W[(k + 1) * HID + n]);
}

// z[n][c] = h_prev[n][c]   (self term of GIN, lda-strided source)
__global__ void copy_rows_kernel(const float* __restrict__ src, int lda, float* __restrict__ dst) {
  int tid = blockIdx.x * blockDim.x + threadIdx.x;
  int n = tid >> 5, j = tid & 31;
  if (n >= N_NODES) return;
  float4 v = *(const float4*)(src + (long long)n * lda + 4 * j);
  *(float4*)(dst + (long long)n * HID + 4 * j) = v;
}

// z[dst] += h[src] over all edges; one lane handles 4 floats.
// 100000x128 f32 node table = 51MB -> lives in the 192MB L2, so the random
// gathers and the 205M atomic adds per layer are L2-resident.
__global__ void edge_scatter_kernel(const float* __restrict__ h, int lda,
                                    const long long* __restrict__ ei,
                                    float* __restrict__ z) {
  long long tid = (long long)blockIdx.x * blockDim.x + threadIdx.x;
  long long e = tid >> 5;
  int j = (int)(tid & 31);
  if (e >= N_EDGES) return;
  long long s = ei[e];
  long long d = ei[N_EDGES + e];
  float4 v = *(const float4*)(h + s * (long long)lda + 4 * j);
  float* zp = z + d * (long long)HID + 4 * j;
  atomicAdd(zp + 0, v.x); atomicAdd(zp + 1, v.y);
  atomicAdd(zp + 2, v.z); atomicAdd(zp + 3, v.w);
}

__global__ void zero_kernel(float* __restrict__ p, int n) {
  int t = blockIdx.x * blockDim.x + threadIdx.x;
  if (t < n) p[t] = 0.0f;
}

// pooled[batch[n]] += feats[n]; block = one node, 384 threads
__global__ void pool_kernel(const float* __restrict__ feats, const long long* __restrict__ batch,
                            float* __restrict__ pooled) {
  int n = blockIdx.x, c = threadIdx.x;
  long long g = batch[n];
  atomicAdd(&pooled[g * (3 * HID) + c], feats[(long long)n * (3 * HID) + c]);
}

// C = act(A[M x K] @ W[K x 128] + bias), N fixed = 128, K compile-time.
// Block = 256 threads = 8 waves owning 32 rows x 128 cols:
//   wave w -> cols [16w, 16w+16), two 16-row m-tiles (B frag reused 2x).
// Whole 32 x K A panel staged to LDS as bf16 once, single barrier, then the
// fully unrolled K loop issues WMMAs back-to-back.
template <int K>
__global__ __launch_bounds__(256) void wmma_gemm_kernel(
    const float* __restrict__ A, int lda,
    const uint32_t* __restrict__ Bp,      // packed fragments, see pack_w_kernel
    const float* __restrict__ bias,       // [128]
    float* __restrict__ C, int ldc, int coff, int relu) {
  constexpr int KD = K / 2;               // dwords (bf16 pairs) per row
  __shared__ uint32_t sA[32 * KD];

  const int t     = threadIdx.x;
  const int lane  = t & 31;
  const int w     = t >> 5;               // column tile 0..7
  const int mbase = blockIdx.x * 32;
  const int m     = lane & 15;
  const int half  = lane >> 4;
  const int n     = w * 16 + m;           // output column 0..127

  // cooperative stage: 32 rows x KD dwords, f32 -> packed bf16
#pragma unroll
  for (int idx = t; idx < 32 * KD; idx += 256) {
    int row = idx / KD, cp = idx % KD;
    float2 f = *(const float2*)(A + (long long)(mbase + row) * lda + 2 * cp);
    sA[idx] = pack2bf(f.x, f.y);
  }
  __syncthreads();

  const float bv = bias[n];
  v8f acc0 = {}, acc1 = {};

#pragma unroll
  for (int kc = 0; kc < K / 32; ++kc) {
    // B fragment: 8 contiguous dwords per lane -> two b128 loads
    Frag16 bfr;
    const uint4* bp = (const uint4*)(Bp + ((((size_t)kc * 8 + w) * 32 + lane) << 3));
    bfr.q[0] = bp[0];
    bfr.q[1] = bp[1];

    // A fragments: 8 contiguous dwords, k = 16*half + s (matches B packing)
    const int kbase = kc * 16 + (half << 3);
    const uint32_t* p0 = sA + (size_t)m * KD + kbase;         // m-tile 0
    const uint32_t* p1 = sA + (size_t)(16 + m) * KD + kbase;  // m-tile 1
    Frag16 a0, a1;
#pragma unroll
    for (int j = 0; j < 4; ++j) {
      a0.u[j] = p0[j];  a0.u[4 + j] = p0[4 + j];
      a1.u[j] = p1[j];  a1.u[4 + j] = p1[4 + j];
    }

    acc0 = __builtin_amdgcn_wmma_f32_16x16x32_bf16(
        false, a0.v, false, bfr.v, (short)0, acc0, false, false);
    acc1 = __builtin_amdgcn_wmma_f32_16x16x32_bf16(
        false, a1.v, false, bfr.v, (short)0, acc1, false, false);
  }

  // C/D layout: VGPR r -> row = 8*half + r within tile, col = lane&15
#pragma unroll
  for (int r = 0; r < 8; ++r) {
    int row0 = mbase + half * 8 + r;
    int row1 = row0 + 16;
    float v0 = acc0[r] + bv;
    float v1 = acc1[r] + bv;
    if (relu) { v0 = fmaxf(v0, 0.0f); v1 = fmaxf(v1, 0.0f); }
    C[(long long)row0 * ldc + coff + n] = v0;
    C[(long long)row1 * ldc + coff + n] = v1;
  }
}

extern "C" void kernel_launch(void* const* d_in, const int* in_sizes, int n_in,
                              void* d_out, int out_size, void* d_ws, size_t ws_size,
                              hipStream_t stream) {
  (void)in_sizes; (void)n_in; (void)out_size; (void)ws_size;

  const float* x   = (const float*)d_in[0];
  const float* W1s = (const float*)d_in[1];
  const float* b1s = (const float*)d_in[2];
  const float* W2s = (const float*)d_in[3];
  const float* b2s = (const float*)d_in[4];
  const float* lW1 = (const float*)d_in[5];
  const float* lb1 = (const float*)d_in[6];
  const float* lW2 = (const float*)d_in[7];
  const float* lb2 = (const float*)d_in[8];
  const long long* ei    = (const long long*)d_in[9];
  const long long* batch = (const long long*)d_in[10];
  float* out = (float*)d_out;

  char* ws = (char*)d_ws;
  size_t off = 0;
  auto alloc = [&](size_t bytes) -> void* {
    void* p = ws + off;
    off += (bytes + 255) & ~(size_t)255;
    return p;
  };
  float*    z      = (float*)alloc((size_t)N_NODES * HID * 4);
  float*    hid    = (float*)alloc((size_t)N_NODES * HID * 4);
  float*    feats  = (float*)alloc((size_t)N_NODES * 3 * HID * 4);
  float*    pooled = (float*)alloc((size_t)N_GRAPHS * 3 * HID * 4);
  float*    phid   = (float*)alloc((size_t)N_GRAPHS * HID * 4);
  uint32_t* W1p    = (uint32_t*)alloc((size_t)N_LAYERS * 64 * HID * 4);
  uint32_t* W2p    = (uint32_t*)alloc((size_t)N_LAYERS * 64 * HID * 4);
  uint32_t* lW1p   = (uint32_t*)alloc((size_t)192 * HID * 4);
  uint32_t* lW2p   = (uint32_t*)alloc((size_t)64 * HID * 4);

  // pack all weights to bf16 fragment layout (tiny, stays hot in L2)
  for (int i = 0; i < N_LAYERS; ++i) {
    pack_w_kernel<<<(64 * HID + 255) / 256, 256, 0, stream>>>(
        W1s + (size_t)i * HID * HID, W1p + (size_t)i * 64 * HID, HID);
    pack_w_kernel<<<(64 * HID + 255) / 256, 256, 0, stream>>>(
        W2s + (size_t)i * HID * HID, W2p + (size_t)i * 64 * HID, HID);
  }
  pack_w_kernel<<<(192 * HID + 255) / 256, 256, 0, stream>>>(lW1, lW1p, 3 * HID);
  pack_w_kernel<<<(64 * HID + 255) / 256, 256, 0, stream>>>(lW2, lW2p, HID);

  const float* hprev = x;
  int lda = HID;
  const long long edge_threads = (long long)N_EDGES * 32;

  for (int i = 0; i < N_LAYERS; ++i) {
    copy_rows_kernel<<<(N_NODES * 32 + 255) / 256, 256, 0, stream>>>(hprev, lda, z);
    edge_scatter_kernel<<<(unsigned)((edge_threads + 255) / 256), 256, 0, stream>>>(hprev, lda, ei, z);
    // hid = ReLU(z @ W1 + b1)
    wmma_gemm_kernel<HID><<<N_NODES / 32, 256, 0, stream>>>(
        z, HID, W1p + (size_t)i * 64 * HID, b1s + i * HID, hid, HID, 0, 1);
    // feats[:, i*128:(i+1)*128] = hid @ W2 + b2
    wmma_gemm_kernel<HID><<<N_NODES / 32, 256, 0, stream>>>(
        hid, HID, W2p + (size_t)i * 64 * HID, b2s + i * HID, feats, 3 * HID, i * HID, 0);
    hprev = feats + i * HID;
    lda = 3 * HID;
  }

  zero_kernel<<<(N_GRAPHS * 3 * HID + 255) / 256, 256, 0, stream>>>(pooled, N_GRAPHS * 3 * HID);
  pool_kernel<<<N_NODES, 3 * HID, 0, stream>>>(feats, batch, pooled);

  // readout MLP: phid = ReLU(pooled @ lW1 + lb1); out = phid @ lW2 + lb2
  wmma_gemm_kernel<3 * HID><<<N_GRAPHS / 32, 256, 0, stream>>>(
      pooled, 3 * HID, lW1p, lb1, phid, HID, 0, 1);
  wmma_gemm_kernel<HID><<<N_GRAPHS / 32, 256, 0, stream>>>(
      phid, HID, lW2p, lb2, out, HID, 0, 0);
}
